// TriangleLinearAttention_5789615915157
// MI455X (gfx1250) — compile-verified, matrix operand order checked by reading, not run on hardware
//
#include <hip/hip_runtime.h>
#include <hip/hip_bf16.h>

// ---------------- types for WMMA ----------------
typedef __bf16 v16bf __attribute__((ext_vector_type(16)));
typedef float  v8f   __attribute__((ext_vector_type(8)));
typedef int    v4i_t __attribute__((ext_vector_type(4)));

union ABf16 { uint4 q[2]; v16bf v; };
union PK8   { unsigned short s[8]; uint4 q; };

#define SEQ   256
#define CZ    128
#define NH    4
#define HD    32
#define FDIM  64
#define NTOK  (SEQ*SEQ)

#define GLOBAL_AS __attribute__((address_space(1)))
#define LDS_AS    __attribute__((address_space(3)))

#if __has_builtin(__builtin_amdgcn_global_load_async_to_lds_b128) && \
    __has_builtin(__builtin_amdgcn_s_wait_asynccnt)
#define HAVE_ASYNC 1
#else
#define HAVE_ASYNC 0
#endif

__device__ __forceinline__ unsigned short f2bf(float x) {
  unsigned int u = __builtin_bit_cast(unsigned int, x);
  unsigned int r = u + 0x7FFFu + ((u >> 16) & 1u);
  return (unsigned short)(r >> 16);
}
__device__ __forceinline__ float bf2f(unsigned short h) {
  unsigned int u = ((unsigned int)h) << 16;
  return __builtin_bit_cast(float, u);
}
__device__ __forceinline__ float clip8(float x) {
  return x < -8.f ? -8.f : (x > 8.f ? 8.f : x);
}
__device__ __forceinline__ float fast_sigmoid(float x) {
  return __builtin_amdgcn_rcpf(1.f + __expf(-x));
}

// Stage N uint4 (16-byte) elements from global to LDS with all 256 threads.
// Uses CDNA5 async copy (ASYNCcnt-tracked, global_load_async_to_lds_b128)
// when the toolchain exposes it.
template <int N>
__device__ __forceinline__ void stage_lds(unsigned short* lds,
                                          const unsigned short* src, int tid) {
#if HAVE_ASYNC
  const GLOBAL_AS v4i_t* gc = (const GLOBAL_AS v4i_t*)src;
  GLOBAL_AS v4i_t* g = const_cast<GLOBAL_AS v4i_t*>(gc);
  LDS_AS v4i_t* l = (LDS_AS v4i_t*)lds;
  for (int idx = tid; idx < N; idx += 256)
    __builtin_amdgcn_global_load_async_to_lds_b128(g + idx, l + idx, 0, 0);
#else
  const uint4* s = (const uint4*)src;
  uint4* d = (uint4*)lds;
  for (int idx = tid; idx < N; idx += 256) d[idx] = s[idx];
#endif
}
__device__ __forceinline__ void stage_wait() {
#if HAVE_ASYNC
  __builtin_amdgcn_s_wait_asynccnt(0);
#endif
}

// ================= Kernel A: weight prep =================
__global__ __launch_bounds__(128) void k_prep(
    const float* __restrict__ q_w, const float* __restrict__ q_b,
    const float* __restrict__ k_w, const float* __restrict__ k_b,
    const float* __restrict__ v_w, const float* __restrict__ v_b,
    const float* __restrict__ qf_w, const float* __restrict__ qf_b,
    const float* __restrict__ kf_w, const float* __restrict__ kf_b,
    const float* __restrict__ g_w, const float* __restrict__ g_b,
    const float* __restrict__ o_w,
    unsigned short* __restrict__ Wbig, float* __restrict__ bias_big,
    unsigned short* __restrict__ oWb)
{
  const int r = blockIdx.x;   // 0..511
  const int c = threadIdx.x;  // 0..127
  float w = 0.f, bias = 0.f;
  if (r < 128) {
    const int h = r >> 5, j = r & 31;
    float s = 0.f, sb = 0.f;
    for (int d = 0; d < 32; ++d) {
      const float f = qf_w[j * 32 + d];
      s  += f * q_w[(h * 32 + d) * CZ + c];
      sb += f * q_b[h * 32 + d];
    }
    w = s; bias = sb + qf_b[j];
  } else if (r < 256) {
    const int h = (r - 128) >> 5, j = (r - 128) & 31;
    float s = 0.f, sb = 0.f;
    for (int d = 0; d < 32; ++d) {
      const float f = kf_w[j * 32 + d];
      s  += f * k_w[(h * 32 + d) * CZ + c];
      sb += f * k_b[h * 32 + d];
    }
    w = s; bias = sb + kf_b[j];
  } else if (r < 384) {
    const int rr = r - 256;
    w = v_w[rr * CZ + c]; bias = v_b[rr];
  } else {
    const int rr = r - 384;
    w = g_w[rr * CZ + c]; bias = g_b[rr];
  }
  Wbig[r * CZ + c] = f2bf(w);
  if (c == 0) bias_big[r] = bias;
  if (r < 128) oWb[r * CZ + c] = f2bf(o_w[r * CZ + c]);
}

// ================= Kernel B: LayerNorm -> bf16 =================
__global__ __launch_bounds__(256) void k_ln(
    const float* __restrict__ z, const float* __restrict__ ln_w,
    const float* __restrict__ ln_b, unsigned short* __restrict__ zn)
{
  const int wave = threadIdx.x >> 5, lane = threadIdx.x & 31;
  const int row = blockIdx.x * 8 + wave;
  const float* zr = z + (size_t)row * CZ;
  const int c = lane * 4;
  float4 v = *(const float4*)(zr + c);
  float s = v.x + v.y + v.z + v.w;
  for (int off = 16; off; off >>= 1) s += __shfl_xor(s, off, 32);
  const float mu = s * (1.f / 128.f);
  const float d0 = v.x - mu, d1 = v.y - mu, d2 = v.z - mu, d3 = v.w - mu;
  float q = d0 * d0 + d1 * d1 + d2 * d2 + d3 * d3;
  for (int off = 16; off; off >>= 1) q += __shfl_xor(q, off, 32);
  const float inv = rsqrtf(q * (1.f / 128.f) + 1e-5f);
  float4 w4 = *(const float4*)(ln_w + c);
  float4 b4 = *(const float4*)(ln_b + c);
  const unsigned int o0 = f2bf(d0 * inv * w4.x + b4.x);
  const unsigned int o1 = f2bf(d1 * inv * w4.y + b4.y);
  const unsigned int o2 = f2bf(d2 * inv * w4.z + b4.z);
  const unsigned int o3 = f2bf(d3 * inv * w4.w + b4.w);
  uint2 t; t.x = o0 | (o1 << 16); t.y = o2 | (o3 << 16);
  *(uint2*)&zn[(size_t)row * CZ + c] = t;
}

// ================= Kernel C: big projection GEMM + epilogues =================
__global__ __launch_bounds__(256) void k_proj(
    const unsigned short* __restrict__ zn,
    const unsigned short* __restrict__ Wbig,
    const float* __restrict__ bias_big,
    const int* __restrict__ mask,
    unsigned short* __restrict__ qf,   // [NTOK][256] = [tok][h*64+f]
    unsigned short* __restrict__ kfT,  // [(i*4+h)*64+f][256]
    unsigned short* __restrict__ vT,   // [(i*4+h)*32+d][256]
    unsigned short* __restrict__ gT)   // [(i*4+h)*32+d][256]
{
  __shared__ unsigned short lA[64 * CZ];   // 16 KB
  __shared__ unsigned short lB[128 * CZ];  // 32 KB
  __shared__ float lBias[512];
  __shared__ float lM[64];
  const int tid = threadIdx.x;
  const int wave = tid >> 5, lane = tid & 31;
  const int m0 = blockIdx.x * 64;

  stage_lds<1024>(lA, zn + (size_t)m0 * CZ, tid);
  for (int idx = tid; idx < 512; idx += 256) lBias[idx] = bias_big[idx];
  if (tid < 64) lM[tid] = mask[m0 + tid] ? 1.f : 0.f;

#pragma unroll
  for (int chunk = 0; chunk < 4; ++chunk) {
    stage_wait();
    __syncthreads();
    stage_lds<2048>(lB, Wbig + (size_t)chunk * 128 * CZ, tid);
    if (chunk < 3)  // hint next chunk into cache (global_prefetch_b8)
      __builtin_prefetch(Wbig + (size_t)(chunk + 1) * 128 * CZ + tid * 64, 0, 0);
    stage_wait();
    __syncthreads();

#pragma unroll
    for (int tt = 0; tt < 4; ++tt) {
      const int t = wave * 4 + tt;
      const int mt = t >> 3, nt = t & 7;
      const int colL = nt * 16 + (lane & 15);   // 0..127
      const int rowA = mt * 16 + (lane & 15);
      const int khalf = (lane >> 4) * 8;
      const float bb = lBias[chunk * 128 + colL];
      v8f acc;
#pragma unroll
      for (int r = 0; r < 8; ++r) acc[r] = bb;
#pragma unroll
      for (int ks = 0; ks < 4; ++ks) {
        const int k0 = ks * 32 + khalf;
        ABf16 a, b;
        a.q[0] = *(const uint4*)&lA[rowA * CZ + k0];
        a.q[1] = *(const uint4*)&lA[rowA * CZ + k0 + 16];
        b.q[0] = *(const uint4*)&lB[colL * CZ + k0];
        b.q[1] = *(const uint4*)&lB[colL * CZ + k0 + 16];
        acc = __builtin_amdgcn_wmma_f32_16x16x32_bf16(
            false, a.v, false, b.v, (short)0, acc, false, false);
      }
      const int mlocBase = mt * 16 + ((lane >> 4) ? 8 : 0);  // local token base
      const int mtokBase = m0 + mlocBase;
      const int iRow = mtokBase >> 8;
      const int jBase = mtokBase & 255;
      const int h = colL >> 5, d = colL & 31;
      if (chunk == 0) {            // qf = [exp(clip(m)), exp(-clip(m))]
#pragma unroll
        for (int r = 0; r < 8; ++r) {
          const float m = clip8(acc[r]);
          const size_t base = (size_t)(mtokBase + r) * 256 + h * 64 + d;
          qf[base]      = f2bf(__expf(m));
          qf[base + 32] = f2bf(__expf(-m));
        }
      } else if (chunk == 1) {     // kf (masked), stored transposed
        PK8 pe, pn;
#pragma unroll
        for (int r = 0; r < 8; ++r) {
          const float mk = lM[mlocBase + r];
          const float m = clip8(acc[r]);
          pe.s[r] = f2bf(__expf(m) * mk);
          pn.s[r] = f2bf(__expf(-m) * mk);
        }
        const size_t b1 = ((size_t)(iRow * NH + h) * FDIM + d) * SEQ + jBase;
        const size_t b2 = ((size_t)(iRow * NH + h) * FDIM + 32 + d) * SEQ + jBase;
        *(uint4*)&kfT[b1] = pe.q;
        *(uint4*)&kfT[b2] = pn.q;
      } else if (chunk == 2) {     // v (masked), stored transposed
        PK8 p;
#pragma unroll
        for (int r = 0; r < 8; ++r) p.s[r] = f2bf(acc[r] * lM[mlocBase + r]);
        const size_t bb2 = ((size_t)(iRow * NH + h) * HD + d) * SEQ + jBase;
        *(uint4*)&vT[bb2] = p.q;
      } else {                     // gate = sigmoid, stored transposed
        PK8 p;
#pragma unroll
        for (int r = 0; r < 8; ++r) p.s[r] = f2bf(fast_sigmoid(acc[r]));
        const size_t bb2 = ((size_t)(iRow * NH + h) * HD + d) * SEQ + jBase;
        *(uint4*)&gT[bb2] = p.q;
      }
    }
  }
}

// ================= Kernel D: linear attention per (row i, head h) =================
__global__ __launch_bounds__(256) void k_attn(
    const unsigned short* __restrict__ qf,
    const unsigned short* __restrict__ kfT,
    const unsigned short* __restrict__ vT,
    const unsigned short* __restrict__ gT,
    unsigned short* __restrict__ attn_g)   // [NTOK][128] bf16 = gate*out
{
  __shared__ unsigned short lK[FDIM * SEQ];  // 32 KB  kfT head
  __shared__ unsigned short lV[HD * SEQ];    // 16 KB  vT head
  __shared__ unsigned short lKV[HD * FDIM];  // 4 KB   kv^T [d][f]
  __shared__ float lSum[FDIM];
  __shared__ float lRDen[SEQ];               // reciprocal of denom
  const int tid = threadIdx.x;
  const int wave = tid >> 5, lane = tid & 31;
  const int bh = blockIdx.x;          // i*4 + h
  const int i = bh >> 2, h = bh & 3;

  stage_lds<2048>(lK, kfT + (size_t)bh * FDIM * SEQ, tid);
  stage_lds<1024>(lV, vT + (size_t)bh * HD * SEQ, tid);
  stage_wait();
  __syncthreads();

  { // kv[f][d] = sum_n kf[n][f] * v[n][d]; one 16x16 tile per wave, K=256
    const int mt = wave >> 1, nt = wave & 1;
    const int fRow = mt * 16 + (lane & 15);
    const int dRow = nt * 16 + (lane & 15);
    const int khalf = (lane >> 4) * 8;
    v8f acc;
#pragma unroll
    for (int r = 0; r < 8; ++r) acc[r] = 0.f;
#pragma unroll
    for (int ks = 0; ks < 8; ++ks) {
      const int k0 = ks * 32 + khalf;
      ABf16 a, b;
      a.q[0] = *(const uint4*)&lK[fRow * SEQ + k0];
      a.q[1] = *(const uint4*)&lK[fRow * SEQ + k0 + 16];
      b.q[0] = *(const uint4*)&lV[dRow * SEQ + k0];
      b.q[1] = *(const uint4*)&lV[dRow * SEQ + k0 + 16];
      acc = __builtin_amdgcn_wmma_f32_16x16x32_bf16(
          false, a.v, false, b.v, (short)0, acc, false, false);
    }
    PK8 p;
#pragma unroll
    for (int r = 0; r < 8; ++r) p.s[r] = f2bf(acc[r]);
    const int fBase = mt * 16 + ((lane >> 4) ? 8 : 0);
    *(uint4*)&lKV[dRow * FDIM + fBase] = p.q;   // transposed: B-layout for stage 2
  }
  __syncthreads();

  if (tid < FDIM) {                    // k_sum[f], 8 bf16 at a time
    float s = 0.f;
    for (int n = 0; n < SEQ; n += 8) {
      PK8 p; p.q = *(const uint4*)&lK[tid * SEQ + n];
#pragma unroll
      for (int e = 0; e < 8; ++e) s += bf2f(p.s[e]);
    }
    lSum[tid] = s;
  }
  __syncthreads();

  { // reciprocal denom per token
    const unsigned short* qrow = qf + ((size_t)(i * SEQ + tid) * 256 + h * FDIM);
    float s = 0.f;
#pragma unroll
    for (int f0 = 0; f0 < FDIM; f0 += 8) {
      PK8 p; p.q = *(const uint4*)(qrow + f0);
#pragma unroll
      for (int e = 0; e < 8; ++e) s += bf2f(p.s[e]) * lSum[f0 + e];
    }
    lRDen[tid] = __builtin_amdgcn_rcpf(s > 1e-6f ? s : 1e-6f);
  }
  __syncthreads();

  // out[n][d] = qf[n][:] @ kv[:][d] * rdenom ; fuse gate multiply
#pragma unroll
  for (int tt = 0; tt < 4; ++tt) {
    const int t = wave * 4 + tt;
    const int mt = t >> 1, nt = t & 1;
    const int rowL = mt * 16 + (lane & 15);
    const int dCol = nt * 16 + (lane & 15);
    const int khalf = (lane >> 4) * 8;
    v8f acc;
#pragma unroll
    for (int r = 0; r < 8; ++r) acc[r] = 0.f;
#pragma unroll
    for (int ks = 0; ks < 2; ++ks) {
      const int k0 = ks * 32 + khalf;
      ABf16 a, b;
      const unsigned short* qrow = qf + ((size_t)(i * SEQ + rowL) * 256 + h * FDIM + k0);
      a.q[0] = *(const uint4*)qrow;
      a.q[1] = *(const uint4*)(qrow + 16);
      b.q[0] = *(const uint4*)&lKV[dCol * FDIM + k0];
      b.q[1] = *(const uint4*)&lKV[dCol * FDIM + k0 + 16];
      acc = __builtin_amdgcn_wmma_f32_16x16x32_bf16(
          false, a.v, false, b.v, (short)0, acc, false, false);
    }
    const int tokBase = mt * 16 + ((lane >> 4) ? 8 : 0);
    PK8 g; g.q = *(const uint4*)&gT[((size_t)bh * HD + dCol) * SEQ + tokBase];
#pragma unroll
    for (int r = 0; r < 8; ++r) {
      const int tokL = tokBase + r;
      const float o = acc[r] * lRDen[tokL];
      attn_g[(size_t)(i * SEQ + tokL) * CZ + h * HD + dCol] = f2bf(bf2f(g.s[r]) * o);
    }
  }
}

// ================= Kernel F: output projection + mask =================
__global__ __launch_bounds__(256) void k_out(
    const unsigned short* __restrict__ attn_g,
    const unsigned short* __restrict__ oWb,
    const float* __restrict__ o_b,
    const int* __restrict__ mask,
    float* __restrict__ out)
{
  __shared__ unsigned short lA[64 * CZ];   // 16 KB
  __shared__ unsigned short lB[128 * CZ];  // 32 KB
  __shared__ float lBias[CZ];
  __shared__ float lM[64];
  const int tid = threadIdx.x;
  const int wave = tid >> 5, lane = tid & 31;
  const int m0 = blockIdx.x * 64;
  stage_lds<1024>(lA, attn_g + (size_t)m0 * CZ, tid);
  stage_lds<2048>(lB, oWb, tid);
  if (tid < CZ) lBias[tid] = o_b[tid];
  if (tid < 64) lM[tid] = mask[m0 + tid] ? 1.f : 0.f;
  stage_wait();
  __syncthreads();

#pragma unroll
  for (int tt = 0; tt < 4; ++tt) {
    const int t = wave * 4 + tt;
    const int mt = t >> 3, nt = t & 7;
    const int colL = nt * 16 + (lane & 15);
    const int rowA = mt * 16 + (lane & 15);
    const int khalf = (lane >> 4) * 8;
    const float bb = lBias[colL];
    v8f acc;
#pragma unroll
    for (int r = 0; r < 8; ++r) acc[r] = bb;
#pragma unroll
    for (int ks = 0; ks < 4; ++ks) {
      const int k0 = ks * 32 + khalf;
      ABf16 a, b;
      a.q[0] = *(const uint4*)&lA[rowA * CZ + k0];
      a.q[1] = *(const uint4*)&lA[rowA * CZ + k0 + 16];
      b.q[0] = *(const uint4*)&lB[colL * CZ + k0];
      b.q[1] = *(const uint4*)&lB[colL * CZ + k0 + 16];
      acc = __builtin_amdgcn_wmma_f32_16x16x32_bf16(
          false, a.v, false, b.v, (short)0, acc, false, false);
    }
    const int mlocBase = mt * 16 + ((lane >> 4) ? 8 : 0);
#pragma unroll
    for (int r = 0; r < 8; ++r) {
      const int tok = m0 + mlocBase + r;
      out[(size_t)tok * CZ + colL] = acc[r] * lM[mlocBase + r];
    }
  }
}

// ================= launch =================
extern "C" void kernel_launch(void* const* d_in, const int* in_sizes, int n_in,
                              void* d_out, int out_size, void* d_ws, size_t ws_size,
                              hipStream_t stream) {
  (void)in_sizes; (void)n_in; (void)out_size; (void)ws_size;
  const float* z    = (const float*)d_in[0];
  const int*   pm   = (const int*)d_in[1];
  const float* ln_w = (const float*)d_in[2];
  const float* ln_b = (const float*)d_in[3];
  const float* q_w  = (const float*)d_in[4];
  const float* q_b  = (const float*)d_in[5];
  const float* k_w  = (const float*)d_in[6];
  const float* k_b  = (const float*)d_in[7];
  const float* v_w  = (const float*)d_in[8];
  const float* v_b  = (const float*)d_in[9];
  const float* qf_w = (const float*)d_in[10];
  const float* qf_b = (const float*)d_in[11];
  const float* kf_w = (const float*)d_in[12];
  const float* kf_b = (const float*)d_in[13];
  const float* g_w  = (const float*)d_in[14];
  const float* g_b  = (const float*)d_in[15];
  const float* o_w  = (const float*)d_in[16];
  const float* o_b  = (const float*)d_in[17];

  char* ws = (char*)d_ws;
  size_t off = 0;
  auto alloc = [&](size_t bytes) { size_t o = off; off += (bytes + 255) & ~(size_t)255; return o; };
  unsigned short* zn     = (unsigned short*)(ws + alloc((size_t)NTOK * CZ * 2));
  unsigned short* Wbig   = (unsigned short*)(ws + alloc(512 * CZ * 2));
  float*          biasB  = (float*)(ws + alloc(512 * 4));
  unsigned short* oWb    = (unsigned short*)(ws + alloc(CZ * CZ * 2));
  unsigned short* qfB    = (unsigned short*)(ws + alloc((size_t)NTOK * 256 * 2));
  unsigned short* kfT    = (unsigned short*)(ws + alloc((size_t)SEQ * NH * FDIM * SEQ * 2));
  unsigned short* vT     = (unsigned short*)(ws + alloc((size_t)SEQ * NH * HD * SEQ * 2));
  unsigned short* gT     = (unsigned short*)(ws + alloc((size_t)SEQ * NH * HD * SEQ * 2));
  unsigned short* attn_g = (unsigned short*)(ws + alloc((size_t)NTOK * CZ * 2));

  k_prep<<<512, 128, 0, stream>>>(q_w, q_b, k_w, k_b, v_w, v_b, qf_w, qf_b,
                                  kf_w, kf_b, g_w, g_b, o_w, Wbig, biasB, oWb);
  k_ln<<<NTOK / 8, 256, 0, stream>>>(z, ln_w, ln_b, zn);
  k_proj<<<NTOK / 64, 256, 0, stream>>>(zn, Wbig, biasB, pm, qfB, kfT, vT, gT);
  k_attn<<<SEQ * NH, 256, 0, stream>>>(qfB, kfT, vT, gT, attn_g);
  k_out<<<NTOK / 64, 256, 0, stream>>>(attn_g, oWb, o_b, pm, (float*)d_out);
}
